// LSTMForcasting_70282844832237
// MI455X (gfx1250) — matrix operand order repaired and checked by reading, not compiled, not run on hardware
//
#include <hip/hip_runtime.h>
#include <hip/hip_bf16.h>
#include <math.h>

// Problem constants (from the reference)
#define BB   512
#define TT   80
#define FF   32
#define HH   512
#define OUTD 2
#define SEQW 64
#define LAH  16
#define RSF  4   // replace_start
#define REF  6   // replace_end

typedef __attribute__((ext_vector_type(16))) __bf16 v16bf;
typedef __attribute__((ext_vector_type(8)))  __bf16 v8bf;
typedef __attribute__((ext_vector_type(8)))  float  v8f;

// ---------------------------------------------------------------------------
// WMMA fragment loaders (wave32; A and B share a contiguous k order, C/D is the ISA layout)
//
// A (16x32 bf16, M x K): lane L -> M = L&15 ; two contiguous 8-half (16B)
//   runs at kc+(L>=16?16:0) and kc+8+(L>=16?16:0).
// B (32x16 bf16, K x N): lane L -> N = L&15 ; one contiguous 16-half (32B)
//   run at kc+(L>=16?16:0). Since the GEMM is act @ W^T with W stored
//   row-major [N,K], this run is contiguous in memory -> no LDS needed.
// C/D (16x16 f32): lane L -> N = L&15, reg r -> M = r + (L>=16 ? 8 : 0).
// ---------------------------------------------------------------------------
__device__ __forceinline__ v16bf load_a_frag(const __bf16* __restrict__ arow,
                                             int kc, int lhi) {
  v8bf lo = *(const v8bf*)(arow + kc + lhi * 16);
  v8bf hi = *(const v8bf*)(arow + kc + 8 + lhi * 16);
  return __builtin_shufflevector(lo, hi, 0,1,2,3,4,5,6,7,8,9,10,11,12,13,14,15);
}

__device__ __forceinline__ v16bf load_b_frag(const __bf16* __restrict__ wrow,
                                             int kc, int khi) {
  return *(const v16bf*)(wrow + kc + khi * 16);
}

__device__ __forceinline__ float sigmoid_(float v) {
  return 1.0f / (1.0f + __expf(-v));
}

#define WMMA_BF16(a, b, c) \
  __builtin_amdgcn_wmma_f32_16x16x32_bf16(false, (a), false, (b), (short)0, (c), false, false)

// ---------------------------------------------------------------------------
// One GEMM pass accumulating 4 independent 16x16 tiles that share the same
// A fragment (the 4 LSTM gates, or 4 adjacent N-tiles for the head).
// Software-pipelined: iteration kc prefetches kc+32's fragments before its
// 4 WMMAs, so the in-flight loads overlap matrix math instead of hitting an
// immediate s_wait_loadcnt 0x0 (which is what the scheduler produced when
// loads and their consuming WMMA sat in the same iteration).
// ---------------------------------------------------------------------------
template<int K>
__device__ __forceinline__ void gemm4_pass(
    const __bf16* __restrict__ arow,
    const __bf16* __restrict__ w0, const __bf16* __restrict__ w1,
    const __bf16* __restrict__ w2, const __bf16* __restrict__ w3,
    int hi, v8f acc[4])
{
  if constexpr (K <= 32) {
    v16bf a  = load_a_frag(arow, 0, hi);
    v16bf b0 = load_b_frag(w0, 0, hi);
    v16bf b1 = load_b_frag(w1, 0, hi);
    v16bf b2 = load_b_frag(w2, 0, hi);
    v16bf b3 = load_b_frag(w3, 0, hi);
    acc[0] = WMMA_BF16(a, b0, acc[0]);
    acc[1] = WMMA_BF16(a, b1, acc[1]);
    acc[2] = WMMA_BF16(a, b2, acc[2]);
    acc[3] = WMMA_BF16(a, b3, acc[3]);
  } else {
    v16bf a  = load_a_frag(arow, 0, hi);
    v16bf b0 = load_b_frag(w0, 0, hi);
    v16bf b1 = load_b_frag(w1, 0, hi);
    v16bf b2 = load_b_frag(w2, 0, hi);
    v16bf b3 = load_b_frag(w3, 0, hi);
#pragma unroll
    for (int kc = 0; kc < K; kc += 32) {
      const int kn = (kc + 32 < K) ? (kc + 32) : 0;  // last prefetch is dead
      v16bf an  = load_a_frag(arow, kn, hi);
      v16bf b0n = load_b_frag(w0, kn, hi);
      v16bf b1n = load_b_frag(w1, kn, hi);
      v16bf b2n = load_b_frag(w2, kn, hi);
      v16bf b3n = load_b_frag(w3, kn, hi);
      acc[0] = WMMA_BF16(a, b0, acc[0]);
      acc[1] = WMMA_BF16(a, b1, acc[1]);
      acc[2] = WMMA_BF16(a, b2, acc[2]);
      acc[3] = WMMA_BF16(a, b3, acc[3]);
      a = an; b0 = b0n; b1 = b1n; b2 = b2n; b3 = b3n;
    }
  }
}

// ---------------------------------------------------------------------------
// Fused LSTM layer step: gates = A1@W1^T (K1) + A2@W2^T (K2) + bias
//                        [+ rank-2 feedback out_prev @ w_ih0[:,4:6]^T]
// then cell update. Each wave owns one 16x16 (batch x hidden) cell tile and
// all 4 gate accumulators, so the nonlinearity is applied fully in-register.
// Grid: 128 blocks x 256 threads = 1024 waves = (B/16)*(H/16) tiles.
// ---------------------------------------------------------------------------
template<int K1, int K2>
__global__ __launch_bounds__(256, 2) void lstm_layer_kernel(
    const __bf16* __restrict__ A1, const __bf16* __restrict__ W1,
    const __bf16* __restrict__ A2, const __bf16* __restrict__ W2,
    const float*  __restrict__ bias,   // [4H] = b_ih + b_hh
    const float*  __restrict__ fbW,    // w_ih0 fp32 [4H, FF] (feedback cols)
    const float*  __restrict__ fbOut,  // [B, 2] previous prediction or null
    float*        __restrict__ c,      // [B, H] in/out (in place, elementwise)
    __bf16*       __restrict__ hout)   // [B, H] bf16 out
{
  const int wave  = (blockIdx.x * 256 + (int)threadIdx.x) >> 5;
  const int lane  = threadIdx.x & 31;
  const int mtile = wave >> 5;           // 0..31 (batch tile)
  const int htile = wave & 31;           // 0..31 (hidden tile)
  const int mbase = mtile << 4;
  const int hbase = htile << 4;
  const int nloc  = lane & 15;
  const int hi    = lane >> 4;

  v8f acc[4] = {};   // i, f, g, o accumulators (f32)

  gemm4_pass<K1>(A1 + (size_t)(mbase + nloc) * K1,
                 W1 + (size_t)(0 * HH + hbase + nloc) * K1,
                 W1 + (size_t)(1 * HH + hbase + nloc) * K1,
                 W1 + (size_t)(2 * HH + hbase + nloc) * K1,
                 W1 + (size_t)(3 * HH + hbase + nloc) * K1, hi, acc);

  gemm4_pass<K2>(A2 + (size_t)(mbase + nloc) * K2,
                 W2 + (size_t)(0 * HH + hbase + nloc) * K2,
                 W2 + (size_t)(1 * HH + hbase + nloc) * K2,
                 W2 + (size_t)(2 * HH + hbase + nloc) * K2,
                 W2 + (size_t)(3 * HH + hbase + nloc) * K2, hi, acc);

  // Epilogue: bias + optional rank-2 feedback + LSTM cell nonlinearity
  const int col = hbase + nloc;
  const float bi = bias[col];
  const float bf = bias[HH + col];
  const float bg = bias[2 * HH + col];
  const float bo = bias[3 * HH + col];

  float wi0 = 0.f, wi1 = 0.f, wf0 = 0.f, wf1 = 0.f;
  float wg0 = 0.f, wg1 = 0.f, wo0 = 0.f, wo1 = 0.f;
  const bool has_fb = (fbOut != nullptr);
  if (has_fb) {
    wi0 = fbW[(0 * HH + col) * FF + RSF];     wi1 = fbW[(0 * HH + col) * FF + RSF + 1];
    wf0 = fbW[(1 * HH + col) * FF + RSF];     wf1 = fbW[(1 * HH + col) * FF + RSF + 1];
    wg0 = fbW[(2 * HH + col) * FF + RSF];     wg1 = fbW[(2 * HH + col) * FF + RSF + 1];
    wo0 = fbW[(3 * HH + col) * FF + RSF];     wo1 = fbW[(3 * HH + col) * FF + RSF + 1];
  }

#pragma unroll
  for (int r = 0; r < 8; ++r) {
    const int row = mbase + r + hi * 8;
    float gi = acc[0][r] + bi;
    float gf = acc[1][r] + bf;
    float gg = acc[2][r] + bg;
    float go = acc[3][r] + bo;
    if (has_fb) {
      const float o0 = fbOut[row * 2 + 0];
      const float o1 = fbOut[row * 2 + 1];
      gi += o0 * wi0 + o1 * wi1;
      gf += o0 * wf0 + o1 * wf1;
      gg += o0 * wg0 + o1 * wg1;
      go += o0 * wo0 + o1 * wo1;
    }
    const size_t idx = (size_t)row * HH + col;
    const float cn = sigmoid_(gf) * c[idx] + sigmoid_(gi) * tanhf(gg);
    c[idx] = cn;
    hout[idx] = (__bf16)(sigmoid_(go) * tanhf(cn));
  }
}

// ---------------------------------------------------------------------------
// Head FC layer: y = gelu_exact(A @ W^T + b). Each wave computes a 16x64
// strip (A-fragment reused across 4 WMMAs, pipelined like the LSTM kernel).
// Grid: 32 blocks x 256 threads = 256 waves = (512/16)*(512/64).
// ---------------------------------------------------------------------------
__global__ __launch_bounds__(256, 2) void head_fc_kernel(
    const __bf16* __restrict__ A, const __bf16* __restrict__ W,
    const float* __restrict__ bias,
    __bf16* __restrict__ ybf, float* __restrict__ yf)
{
  const int wave   = (blockIdx.x * 256 + (int)threadIdx.x) >> 5;
  const int lane   = threadIdx.x & 31;
  const int mtile  = wave >> 3;          // 0..31
  const int nstrip = wave & 7;           // 0..7
  const int mbase  = mtile << 4;
  const int nbase  = nstrip << 6;
  const int nloc   = lane & 15;
  const int hi     = lane >> 4;

  v8f acc[4] = {};
  gemm4_pass<HH>(A + (size_t)(mbase + nloc) * HH,
                 W + (size_t)(nbase +  0 + nloc) * HH,
                 W + (size_t)(nbase + 16 + nloc) * HH,
                 W + (size_t)(nbase + 32 + nloc) * HH,
                 W + (size_t)(nbase + 48 + nloc) * HH, hi, acc);

#pragma unroll
  for (int j = 0; j < 4; ++j) {
    const int n = nbase + j * 16 + nloc;
    const float bn = bias[n];
#pragma unroll
    for (int r = 0; r < 8; ++r) {
      const int row = mbase + r + hi * 8;
      const float v = acc[j][r] + bn;
      const float g = 0.5f * v * (1.0f + erff(v * 0.70710678118654752f)); // exact GELU
      const size_t idx = (size_t)row * HH + n;
      if (ybf) ybf[idx] = (__bf16)g;
      if (yf)  yf[idx]  = g;
    }
  }
}

// fc3: [B,512] fp32 @ [2,512]^T fp32 + b -> prediction; scatter to d_out and
// write feedback buffer. 4 blocks x 256 = 1024 threads = B*OUTD outputs.
__global__ void fc3_kernel(const float* __restrict__ y,
                           const float* __restrict__ w,
                           const float* __restrict__ b,
                           float* __restrict__ outbuf,
                           float* __restrict__ dout, int s)
{
  const int tid = blockIdx.x * 256 + threadIdx.x;
  const int row = tid >> 1;
  const int j   = tid & 1;
  const float* yr = y + (size_t)row * HH;
  const float* wr = w + (size_t)j * HH;
  float acc = b[j];
  for (int k = 0; k < HH; ++k) acc += yr[k] * wr[k];
  outbuf[row * 2 + j] = acc;
  dout[(size_t)row * ((LAH + 1) * OUTD) + s * OUTD + j] = acc;
}

// ---------------------------------------------------------------------------
// Prep kernels
// ---------------------------------------------------------------------------
__global__ void cvt_bf16_kernel(const float* __restrict__ src,
                                __bf16* __restrict__ dst, int n) {
  const int i = blockIdx.x * 256 + threadIdx.x;
  if (i < n) dst[i] = (__bf16)src[i];
}

__global__ void bias_add_kernel(const float* __restrict__ a,
                                const float* __restrict__ b,
                                float* __restrict__ o, int n) {
  const int i = blockIdx.x * 256 + threadIdx.x;
  if (i < n) o[i] = a[i] + b[i];
}

__global__ void zero_f32_kernel(float* __restrict__ p, int n) {
  const int i = blockIdx.x * 256 + threadIdx.x;
  if (i < n) p[i] = 0.0f;
}

__global__ void zero_bf16_kernel(__bf16* __restrict__ p, int n) {
  const int i = blockIdx.x * 256 + threadIdx.x;
  if (i < n) p[i] = (__bf16)0.0f;
}

// x [B,T,F] fp32 -> xbf [T,B,F] bf16, zeroing features [RS:RE) for t >= SEQ
// (those contributions are re-added exactly via the fp32 feedback term).
__global__ void build_xbf_kernel(const float* __restrict__ x,
                                 __bf16* __restrict__ xbf) {
  const int i = blockIdx.x * 256 + threadIdx.x;   // over T*B*F
  if (i >= TT * BB * FF) return;
  const int f = i % FF;
  const int b = (i / FF) % BB;
  const int t = i / (FF * BB);
  float v = x[((size_t)b * TT + t) * FF + f];
  if (t >= SEQW && f >= RSF && f < REF) v = 0.0f;
  xbf[i] = (__bf16)v;
}

// ---------------------------------------------------------------------------
// Host orchestration: ~224 kernels, all captured into the HIP graph.
// ---------------------------------------------------------------------------
extern "C" void kernel_launch(void* const* d_in, const int* in_sizes, int n_in,
                              void* d_out, int out_size, void* d_ws, size_t ws_size,
                              hipStream_t stream) {
  (void)in_sizes; (void)n_in; (void)out_size; (void)ws_size;
  const float* x      = (const float*)d_in[0];
  const float* w_ih0  = (const float*)d_in[1];
  const float* w_hh0  = (const float*)d_in[2];
  const float* b_ih0  = (const float*)d_in[3];
  const float* b_hh0  = (const float*)d_in[4];
  const float* w_ih1  = (const float*)d_in[5];
  const float* w_hh1  = (const float*)d_in[6];
  const float* b_ih1  = (const float*)d_in[7];
  const float* b_hh1  = (const float*)d_in[8];
  const float* fc1_w  = (const float*)d_in[9];
  const float* fc1_b  = (const float*)d_in[10];
  const float* fc2_w  = (const float*)d_in[11];
  const float* fc2_b  = (const float*)d_in[12];
  const float* fc3_w  = (const float*)d_in[13];
  const float* fc3_b  = (const float*)d_in[14];
  float* out = (float*)d_out;

  char* ws = (char*)d_ws;
  size_t off = 0;
  auto alloc = [&](size_t bytes) -> void* {
    void* p = ws + off;
    off = (off + bytes + 255) & ~(size_t)255;
    return p;
  };

  __bf16* w_hh0bf = (__bf16*)alloc((size_t)4 * HH * HH * 2);
  __bf16* w_ih1bf = (__bf16*)alloc((size_t)4 * HH * HH * 2);
  __bf16* w_hh1bf = (__bf16*)alloc((size_t)4 * HH * HH * 2);
  __bf16* w_ih0bf = (__bf16*)alloc((size_t)4 * HH * FF * 2);
  __bf16* fc1bf   = (__bf16*)alloc((size_t)HH * HH * 2);
  __bf16* fc2bf   = (__bf16*)alloc((size_t)HH * HH * 2);
  float*  bias0   = (float*)alloc((size_t)4 * HH * 4);
  float*  bias1   = (float*)alloc((size_t)4 * HH * 4);
  __bf16* xbf     = (__bf16*)alloc((size_t)TT * BB * FF * 2);
  __bf16* h0A     = (__bf16*)alloc((size_t)BB * HH * 2);
  __bf16* h0B     = (__bf16*)alloc((size_t)BB * HH * 2);
  __bf16* h1A     = (__bf16*)alloc((size_t)BB * HH * 2);
  __bf16* h1B     = (__bf16*)alloc((size_t)BB * HH * 2);
  float*  c0      = (float*)alloc((size_t)BB * HH * 4);
  float*  c1      = (float*)alloc((size_t)BB * HH * 4);
  __bf16* y1bf    = (__bf16*)alloc((size_t)BB * HH * 2);
  float*  y2f     = (float*)alloc((size_t)BB * HH * 4);
  float*  outbuf  = (float*)alloc((size_t)BB * OUTD * 4);

  // ---- prep: weight conversion, fused biases, x transpose, state zero ----
  cvt_bf16_kernel<<<4096, 256, 0, stream>>>(w_hh0, w_hh0bf, 4 * HH * HH);
  cvt_bf16_kernel<<<4096, 256, 0, stream>>>(w_ih1, w_ih1bf, 4 * HH * HH);
  cvt_bf16_kernel<<<4096, 256, 0, stream>>>(w_hh1, w_hh1bf, 4 * HH * HH);
  cvt_bf16_kernel<<<256,  256, 0, stream>>>(w_ih0, w_ih0bf, 4 * HH * FF);
  cvt_bf16_kernel<<<1024, 256, 0, stream>>>(fc1_w, fc1bf, HH * HH);
  cvt_bf16_kernel<<<1024, 256, 0, stream>>>(fc2_w, fc2bf, HH * HH);
  bias_add_kernel<<<8, 256, 0, stream>>>(b_ih0, b_hh0, bias0, 4 * HH);
  bias_add_kernel<<<8, 256, 0, stream>>>(b_ih1, b_hh1, bias1, 4 * HH);
  build_xbf_kernel<<<(TT * BB * FF) / 256, 256, 0, stream>>>(x, xbf);
  zero_bf16_kernel<<<1024, 256, 0, stream>>>(h0A, BB * HH);
  zero_bf16_kernel<<<1024, 256, 0, stream>>>(h1A, BB * HH);
  zero_f32_kernel<<<1024, 256, 0, stream>>>(c0, BB * HH);
  zero_f32_kernel<<<1024, 256, 0, stream>>>(c1, BB * HH);

  // ---- sequential scan: 2 fused WMMA kernels per timestep ----
  for (int t = 0; t < TT; ++t) {
    const __bf16* h0_in  = (t & 1) ? h0B : h0A;
    __bf16*       h0_out = (t & 1) ? h0A : h0B;
    const __bf16* h1_in  = (t & 1) ? h1B : h1A;
    __bf16*       h1_out = (t & 1) ? h1A : h1B;
    const float*  fb     = (t >= SEQW) ? outbuf : nullptr;

    // layer 0: h0 @ w_hh0^T (K=512) + x_t @ w_ih0^T (K=32) + bias0 [+ fb]
    lstm_layer_kernel<HH, FF><<<128, 256, 0, stream>>>(
        h0_in, w_hh0bf, xbf + (size_t)t * BB * FF, w_ih0bf,
        bias0, w_ih0, fb, c0, h0_out);

    // layer 1: h1 @ w_hh1^T (K=512) + h0' @ w_ih1^T (K=512) + bias1
    lstm_layer_kernel<HH, HH><<<128, 256, 0, stream>>>(
        h1_in, w_hh1bf, h0_out, w_ih1bf,
        bias1, nullptr, nullptr, c1, h1_out);

    if (t >= SEQW - 1) {
      const int s = t - (SEQW - 1);   // output slot 0..LA
      head_fc_kernel<<<32, 256, 0, stream>>>(h1_out, fc1bf, fc1_b, y1bf, nullptr);
      head_fc_kernel<<<32, 256, 0, stream>>>(y1bf, fc2bf, fc2_b, nullptr, y2f);
      fc3_kernel<<<4, 256, 0, stream>>>(y2f, fc3_w, fc3_b, outbuf, out, s);
    }
  }
}
